// WuBuNestingModel_54546084659755
// MI455X (gfx1250) — matrix operand order repaired and checked by reading, not compiled, not run on hardware
//
#include <hip/hip_runtime.h>
#include <hip/hip_bf16.h>
#include <math.h>

typedef __attribute__((ext_vector_type(16))) __bf16 v16bf;
typedef __attribute__((ext_vector_type(8)))  float  v8f;

#define EPSV 1e-7f

// v_tan = g(||v||) * v : expmap0 -> projx -> logmap0 chain collapses to a scalar
// gain because every step rescales along the same direction.
__device__ __forceinline__ float hyp_g(float n, float c, float s) {
  float sc = sqrtf(c);
  float nu = s * n;
  float n1 = fmaxf(nu, EPSV);
  float f1 = tanhf(sc * n1) / (sc * n1);      // expmap0 factor (applied to s*v)
  float nh_ = f1 * nu;                         // ||x_hyp||
  float nh = fmaxf(nh_, EPSV);
  float mx = (1.0f - 1e-5f) / sc;
  float f2 = (nh > mx) ? (mx / nh) : 1.0f;     // projx factor
  float n2 = f2 * nh_;
  float n3 = fmaxf(n2, EPSV);
  float a  = fminf(fmaxf(sc * n3, 0.0f), 1.0f - 1e-7f);
  float f3 = atanhf(a) / (sc * n3);            // logmap0 factor
  return f1 * f2 * f3;                         // (s and 1/s cancel)
}

__device__ __forceinline__ float gelu_f(float x) {
  return 0.5f * x * (1.0f + erff(x * 0.7071067811865476f));
}

__device__ __forceinline__ void ham(const float* a, const float* b, float* o) {
  o[0] = a[0]*b[0] - a[1]*b[1] - a[2]*b[2] - a[3]*b[3];
  o[1] = a[0]*b[1] + a[1]*b[0] + a[2]*b[3] - a[3]*b[2];
  o[2] = a[0]*b[2] - a[1]*b[3] + a[2]*b[0] + a[3]*b[1];
  o[3] = a[0]*b[3] + a[1]*b[2] - a[2]*b[1] + a[3]*b[0];
}

// ---------------------------------------------------------------------------
// prep_fuse: Wf = W_tan @ W_in  (64 x 784, zero-padded to K=800) as bf16,
//            bfv = W_tan @ b_in + b_tan
// ---------------------------------------------------------------------------
__global__ __launch_bounds__(256) void prep_fuse(
    const float* __restrict__ W_in, const float* __restrict__ b_in,
    const float* __restrict__ W_tan, const float* __restrict__ b_tan,
    __bf16* __restrict__ Wf, float* __restrict__ bfv) {
  int idx = blockIdx.x * 256 + threadIdx.x;     // 64*800 = 51200 total
  if (idx < 64 * 800) {
    int n = idx / 800, k = idx % 800;
    float acc = 0.0f;
    if (k < 784)
      for (int j = 0; j < 128; ++j) acc += W_tan[n * 128 + j] * W_in[j * 784 + k];
    Wf[idx] = (__bf16)acc;
  }
  if (idx < 64) {
    float acc = b_tan[idx];
    for (int j = 0; j < 128; ++j) acc += W_tan[idx * 128 + j] * b_in[j];
    bfv[idx] = acc;
  }
}

// ---------------------------------------------------------------------------
// prep_small: R = expm(A - A^T) (scaled Taylor-12 + 6 squarings), boundary
// chains, quaternion matrix; emits fused MLP first-layer weights/biases:
//   W0aF = 2*(W0a@R), b0aF = b0a - (W0a@R)@mb0
//   W1aF = 2*(W1a@M), b1aF = b1a - (W1a@M)@mb1
// ---------------------------------------------------------------------------
__global__ __launch_bounds__(256) void prep_small(
    const float* __restrict__ A, const float* __restrict__ bp0,
    const float* __restrict__ bp1, const float* __restrict__ pq,
    const float* __restrict__ qq, const float* __restrict__ W0a,
    const float* __restrict__ b0a, const float* __restrict__ W0b,
    const float* __restrict__ b0b, const float* __restrict__ W1a,
    const float* __restrict__ b1a,
    float* __restrict__ W0aF, float* __restrict__ b0aF,
    float* __restrict__ W1aF, float* __restrict__ b1aF) {
  __shared__ float S[4096];   // generator / later: small scratch
  __shared__ float P[4096];   // accumulates expm, ends as R
  __shared__ float T[4096];   // Taylor term / later: W0a@R
  int tid = threadIdx.x;

  for (int e = tid; e < 4096; e += 256) {
    int i = e >> 6, j = e & 63;
    S[e] = (A[i * 64 + j] - A[j * 64 + i]) * (1.0f / 64.0f);  // G / 2^6
  }
  __syncthreads();
  for (int e = tid; e < 4096; e += 256) {
    int i = e >> 6, j = e & 63;
    P[e] = S[e] + ((i == j) ? 1.0f : 0.0f);
    T[e] = S[e];
  }
  __syncthreads();
  for (int k = 2; k <= 12; ++k) {                 // Taylor terms
    float u[16];
    for (int t = 0; t < 16; ++t) {
      int e = tid + t * 256, i = e >> 6, j = e & 63;
      float acc = 0.0f;
      for (int m = 0; m < 64; ++m) acc += T[i * 64 + m] * S[m * 64 + j];
      u[t] = acc / (float)k;
    }
    __syncthreads();
    for (int t = 0; t < 16; ++t) { int e = tid + t * 256; T[e] = u[t]; P[e] += u[t]; }
    __syncthreads();
  }
  for (int sq = 0; sq < 6; ++sq) {                // squarings: P = P@P
    float u[16];
    for (int t = 0; t < 16; ++t) {
      int e = tid + t * 256, i = e >> 6, j = e & 63;
      float acc = 0.0f;
      for (int m = 0; m < 64; ++m) acc += P[i * 64 + m] * P[m * 64 + j];
      u[t] = acc;
    }
    __syncthreads();
    for (int t = 0; t < 16; ++t) P[tid + t * 256] = u[t];
    __syncthreads();
  }
  // P == R. Now T := W0a @ R   (32 x 64)
  for (int e = tid; e < 2048; e += 256) {
    int o = e >> 6, k = e & 63;
    float acc = 0.0f;
    for (int j = 0; j < 64; ++j) acc += W0a[o * 64 + j] * P[j * 64 + k];
    T[e] = acc;
  }
  // boundary level-0 tangents (c=1, s=1) -> S[0..319]
  if (tid < 5) {
    float n = 0.0f;
    for (int k = 0; k < 64; ++k) { float t = bp0[tid * 64 + k]; n += t * t; }
    float g = hyp_g(sqrtf(n), 1.0f, 1.0f);
    for (int k = 0; k < 64; ++k) S[tid * 64 + k] = g * bp0[tid * 64 + k];
  }
  __syncthreads();
  if (tid < 64) {                                  // mb0 -> S[320..383]
    float acc = 0.0f;
    for (int p = 0; p < 5; ++p) acc += S[p * 64 + tid];
    S[320 + tid] = acc * 0.2f;
  }
  __syncthreads();
  if (tid < 32) {                                  // b0aF
    float acc = b0a[tid];
    for (int k = 0; k < 64; ++k) acc -= T[tid * 64 + k] * S[320 + k];
    b0aF[tid] = acc;
  }
  for (int e = tid; e < 2048; e += 256) W0aF[e] = 2.0f * T[e];
  // carried1 = mlp0(bown0 @ R^T) -> S[384..403]
  if (tid < 5) {
    float br[64];
    for (int k = 0; k < 64; ++k) {
      float acc = 0.0f;
      for (int j = 0; j < 64; ++j) acc += S[tid * 64 + j] * P[k * 64 + j];
      br[k] = acc;
    }
    float t1[32];
    for (int o = 0; o < 32; ++o) {
      float acc = b0a[o];
      for (int j = 0; j < 64; ++j) acc += W0a[o * 64 + j] * br[j];
      t1[o] = gelu_f(acc);
    }
    for (int o = 0; o < 4; ++o) {
      float acc = b0b[o];
      for (int j = 0; j < 32; ++j) acc += W0b[o * 32 + j] * t1[j];
      S[384 + tid * 4 + o] = acc;
    }
  }
  // b_own1 (c=0.8, s=1) -> S[404..415]
  if (tid >= 8 && tid < 11) {
    int p = tid - 8;
    float n = 0.0f;
    for (int k = 0; k < 4; ++k) { float t = bp1[p * 4 + k]; n += t * t; }
    float g = hyp_g(sqrtf(n), 0.8f, 1.0f);
    for (int k = 0; k < 4; ++k) S[404 + p * 4 + k] = g * bp1[p * 4 + k];
  }
  __syncthreads();
  if (tid == 0) {
    for (int j = 0; j < 4; ++j) {                  // mb1 over 8 rows -> S[416..419]
      float acc = 0.0f;
      for (int p = 0; p < 3; ++p) acc += S[404 + p * 4 + j];
      for (int p = 0; p < 5; ++p) acc += S[384 + p * 4 + j];
      S[416 + j] = acc * 0.125f;
    }
    float pn = 0.0f, qn = 0.0f, ph[4], qh[4];
    for (int i = 0; i < 4; ++i) { pn += pq[i] * pq[i]; qn += qq[i] * qq[i]; }
    pn = sqrtf(pn) + 1e-7f; qn = sqrtf(qn) + 1e-7f;
    for (int i = 0; i < 4; ++i) { ph[i] = pq[i] / pn; qh[i] = qq[i] / qn; }
    for (int j = 0; j < 4; ++j) {                  // M[i][j] = (p̂⊗e_j)⊗q̂ -> S[420..435]
      float e[4] = {0, 0, 0, 0}; e[j] = 1.0f;
      float t[4], o[4];
      ham(ph, e, t); ham(t, qh, o);
      for (int i = 0; i < 4; ++i) S[420 + i * 4 + j] = o[i];
    }
  }
  __syncthreads();
  if (tid < 16) {
    float wm[4];
    for (int j = 0; j < 4; ++j) {
      float acc = 0.0f;
      for (int t = 0; t < 4; ++t) acc += W1a[tid * 4 + t] * S[420 + t * 4 + j];
      wm[j] = acc;
    }
    float acc = b1a[tid];
    for (int j = 0; j < 4; ++j) acc -= wm[j] * S[416 + j];
    b1aF[tid] = acc;
    for (int j = 0; j < 4; ++j) W1aF[tid * 4 + j] = 2.0f * wm[j];
  }
}

// ---------------------------------------------------------------------------
// Main: per block 64 rows (4 waves x 16 rows). WMMA bf16 GEMM over K=800
// (25 chunks of 32), then lane-pair-per-row VALU tail.
// ---------------------------------------------------------------------------
__global__ __launch_bounds__(128) void wubu_main(
    const float* __restrict__ x, const __bf16* __restrict__ Wf,
    const float* __restrict__ bfv, const float* __restrict__ W0aF,
    const float* __restrict__ b0aF, const float* __restrict__ W0b,
    const float* __restrict__ b0b, const float* __restrict__ W1aF,
    const float* __restrict__ b1aF, const float* __restrict__ W1b,
    const float* __restrict__ b1b, const float* __restrict__ Wout,
    const float* __restrict__ bout, float* __restrict__ out) {
  __shared__ float sW0a[2048], sb0a[32];
  __shared__ float sW0b[128],  sb0b[4];
  __shared__ float sW1a[64],   sb1a[16];
  __shared__ float sW1b[512],  sb1b[32];
  __shared__ float sWout[1000], sbout[10];
  __shared__ float sbfv[64];
  __shared__ float vbuf[4][16][68];   // v / v_tan0 (stride 68: conflict-free)
  __shared__ float tbuf[4][16][32];   // MLP hidden scratch
  __shared__ float vt1[4][16][4];     // v_tan1
  __shared__ float vt2[4][16][32];    // v_tan2

  int tid = threadIdx.x;
  for (int i = tid; i < 2048; i += 128) sW0a[i] = W0aF[i];
  for (int i = tid; i < 512;  i += 128) sW1b[i] = W1b[i];
  for (int i = tid; i < 1000; i += 128) sWout[i] = Wout[i];
  for (int i = tid; i < 128;  i += 128) sW0b[i] = W0b[i];
  if (tid < 64) { sW1a[tid] = W1aF[tid]; sbfv[tid] = bfv[tid]; }
  if (tid < 32) { sb0a[tid] = b0aF[tid]; sb1b[tid] = b1b[tid]; }
  if (tid < 16) sb1a[tid] = b1aF[tid];
  if (tid < 10) sbout[tid] = bout[tid];
  if (tid < 4)  sb0b[tid] = b0b[tid];
  __syncthreads();

  int lane = tid & 31, w = tid >> 5;
  int r = lane & 15, h = lane >> 4;
  int rowg = blockIdx.x * 64 + w * 16 + r;
  const float* xr = x + (size_t)rowg * 784;

  v8f acc[4] = {v8f{}, v8f{}, v8f{}, v8f{}};
  for (int kc = 0; kc < 25; ++kc) {
    // A fragment, 16x32 bf16, contiguous k order (same as B): lane(h,r): K = h*16+0..15
    v16bf a;
    int k0 = kc * 32 + h * 16;
    if (kc < 24) {
      const float4* p0 = (const float4*)(xr + k0);
      const float4* p1 = (const float4*)(xr + k0 + 8);
      float4 q0 = p0[0], q1 = p0[1], q2 = p1[0], q3 = p1[1];
      a[0]  = (__bf16)q0.x; a[1]  = (__bf16)q0.y; a[2]  = (__bf16)q0.z; a[3]  = (__bf16)q0.w;
      a[4]  = (__bf16)q1.x; a[5]  = (__bf16)q1.y; a[6]  = (__bf16)q1.z; a[7]  = (__bf16)q1.w;
      a[8]  = (__bf16)q2.x; a[9]  = (__bf16)q2.y; a[10] = (__bf16)q2.z; a[11] = (__bf16)q2.w;
      a[12] = (__bf16)q3.x; a[13] = (__bf16)q3.y; a[14] = (__bf16)q3.z; a[15] = (__bf16)q3.w;
    } else {
      for (int t = 0; t < 8; ++t) { int k = k0 + t;      a[t]     = (__bf16)((k < 784) ? xr[k] : 0.0f); }
      for (int t = 0; t < 8; ++t) { int k = k0 + 8 + t; a[8 + t] = (__bf16)((k < 784) ? xr[k] : 0.0f); }
    }
#pragma unroll
    for (int nt = 0; nt < 4; ++nt) {
      // B fragment 32x16: lane(h,r): column N=nt*16+r, K = h*16..h*16+15 (contig bf16)
      v16bf b = *(const v16bf*)(Wf + (size_t)(nt * 16 + r) * 800 + kc * 32 + h * 16);
      acc[nt] = __builtin_amdgcn_wmma_f32_16x16x32_bf16(
          false, a, false, b, (short)0, acc[nt], false, false);
    }
  }
  // bias + C-layout (VGPR j: row h*8+j, col nt*16+r) -> vbuf
#pragma unroll
  for (int nt = 0; nt < 4; ++nt) {
    float bias = sbfv[nt * 16 + r];
#pragma unroll
    for (int j = 0; j < 8; ++j)
      vbuf[w][h * 8 + j][nt * 16 + r] = acc[nt][j] + bias;
  }

  // ---- tail: lane pair (r, r+16) per row, h splits the work ----
  // level 0 hyperbolic (c=1,s=1); scale v in place -> v_tan0
  float ps = 0.0f;
  for (int k = 0; k < 32; ++k) { float t = vbuf[w][r][h * 32 + k]; ps += t * t; }
  ps += __shfl_xor(ps, 16, 32);
  float g0 = hyp_g(sqrtf(ps), 1.0f, 1.0f);
  for (int k = 0; k < 32; ++k) vbuf[w][r][h * 32 + k] *= g0;

  // mlp0 layer1 (W0aF = 2*W0a@R, b0aF folds -mb0): 16 outputs per lane
  for (int t = 0; t < 16; ++t) {
    int o = h * 16 + t;
    float a0 = sb0a[o];
    for (int k = 0; k < 64; ++k) a0 += sW0a[o * 64 + k] * vbuf[w][r][k];
    tbuf[w][r][o] = gelu_f(a0);
  }
  // mlp0 layer2 -> v1[4]; level-1 hyperbolic (c=0.8,s=0.8)
  float v1[2];
  for (int t = 0; t < 2; ++t) {
    int o = h * 2 + t;
    float a0 = sb0b[o];
    for (int k = 0; k < 32; ++k) a0 += sW0b[o * 32 + k] * tbuf[w][r][k];
    v1[t] = a0;
  }
  float ps1 = v1[0] * v1[0] + v1[1] * v1[1];
  ps1 += __shfl_xor(ps1, 16, 32);
  float g1 = hyp_g(sqrtf(ps1), 0.8f, 0.8f);
  vt1[w][r][h * 2 + 0] = g1 * v1[0];
  vt1[w][r][h * 2 + 1] = g1 * v1[1];

  // mlp1 layer1 (W1aF = 2*W1a@M, b1aF folds -mb1): 8 outputs per lane
  for (int t = 0; t < 8; ++t) {
    int o = h * 8 + t;
    float a0 = sb1a[o];
    for (int k = 0; k < 4; ++k) a0 += sW1a[o * 4 + k] * vt1[w][r][k];
    tbuf[w][r][o] = gelu_f(a0);
  }
  // mlp1 layer2 -> v2[32]; level-2 hyperbolic (c=1.2,s=0.6)
  float v2[16], ps2 = 0.0f;
  for (int t = 0; t < 16; ++t) {
    int o = h * 16 + t;
    float a0 = sb1b[o];
    for (int k = 0; k < 16; ++k) a0 += sW1b[o * 16 + k] * tbuf[w][r][k];
    v2[t] = a0; ps2 += a0 * a0;
  }
  ps2 += __shfl_xor(ps2, 16, 32);
  float g2 = hyp_g(sqrtf(ps2), 1.2f, 0.6f);
  for (int t = 0; t < 16; ++t) vt2[w][r][h * 16 + t] = g2 * v2[t];

  // output head: concat(v_tan0[64], v_tan1[4], v_tan2[32]) @ W_out^T + b_out
  for (int t = 0; t < 5; ++t) {
    int o = h * 5 + t;
    float a0 = sbout[o];
    for (int k = 0; k < 64; ++k) a0 += sWout[o * 100 + k] * vbuf[w][r][k];
    for (int k = 0; k < 4; ++k)  a0 += sWout[o * 100 + 64 + k] * vt1[w][r][k];
    for (int k = 0; k < 32; ++k) a0 += sWout[o * 100 + 68 + k] * vt2[w][r][k];
    out[(size_t)rowg * 10 + o] = a0;
  }
}

extern "C" void kernel_launch(void* const* d_in, const int* in_sizes, int n_in,
                              void* d_out, int out_size, void* d_ws, size_t ws_size,
                              hipStream_t stream) {
  const float* x     = (const float*)d_in[0];
  const float* W_in  = (const float*)d_in[1];
  const float* b_in  = (const float*)d_in[2];
  const float* W_tan = (const float*)d_in[3];
  const float* b_tan = (const float*)d_in[4];
  const float* bp0   = (const float*)d_in[5];
  const float* bp1   = (const float*)d_in[6];
  /* d_in[7] = bp2 : dead in the reference (v_comb at level 2 unused) */
  const float* A_rot = (const float*)d_in[8];
  const float* p_q   = (const float*)d_in[9];
  const float* q_q   = (const float*)d_in[10];
  const float* W0a   = (const float*)d_in[11];
  const float* b0a   = (const float*)d_in[12];
  const float* W0b   = (const float*)d_in[13];
  const float* b0b   = (const float*)d_in[14];
  const float* W1a   = (const float*)d_in[15];
  const float* b1a   = (const float*)d_in[16];
  const float* W1b   = (const float*)d_in[17];
  const float* b1b   = (const float*)d_in[18];
  const float* W_out = (const float*)d_in[19];
  const float* b_out = (const float*)d_in[20];

  char* ws = (char*)d_ws;
  __bf16* Wf  = (__bf16*)ws;                    // 64*800 bf16 = 102400 B
  float* bfv  = (float*)(ws + 102400);          // 64 f32
  float* W0aF = (float*)(ws + 102656);          // 32*64 f32
  float* b0aF = (float*)(ws + 110848);          // 32 f32
  float* W1aF = (float*)(ws + 110976);          // 16*4 f32
  float* b1aF = (float*)(ws + 111232);          // 16 f32

  int B = in_sizes[0] / 784;                    // 65536

  prep_fuse<<<dim3((64 * 800 + 255) / 256), dim3(256), 0, stream>>>(
      W_in, b_in, W_tan, b_tan, Wf, bfv);
  prep_small<<<dim3(1), dim3(256), 0, stream>>>(
      A_rot, bp0, bp1, p_q, q_q, W0a, b0a, W0b, b0b, W1a, b1a,
      W0aF, b0aF, W1aF, b1aF);
  wubu_main<<<dim3(B / 64), dim3(128), 0, stream>>>(
      x, Wf, bfv, W0aF, b0aF, W0b, b0b, W1aF, b1aF, W1b, b1b,
      W_out, b_out, (float*)d_out);
}
